// LocalBackwardTemporalAttention_90898687852563
// MI455X (gfx1250) — matrix operand order repaired and checked by reading, not compiled, not run on hardware
//
#include <hip/hip_runtime.h>

// ============================================================================
// MI455X (gfx1250, wave32) implementation of LocalBackwardTemporalAttention.
// All GEMMs run on V_WMMA_F32_16X16X32_BF16 with f32 accumulation.
// GEMM main loop uses CDNA5 async memory->LDS copies (ASYNCcnt) with
// double-buffered LDS tiles.
// ============================================================================

typedef __bf16 bf16_t;
typedef __attribute__((ext_vector_type(16))) __bf16 v16bf;
typedef __attribute__((ext_vector_type(8)))  float  v8f;

#define LN_EPS 1e-6f

// ---- CDNA5 async global->LDS copy support (probe via __has_builtin) --------
#if defined(__has_builtin)
#if __has_builtin(__builtin_amdgcn_global_load_async_to_lds_b128)
#define USE_ASYNC_LDS 1
#endif
#endif
#ifndef USE_ASYNC_LDS
#define USE_ASYNC_LDS 0
#endif

#if defined(__has_builtin) && __has_builtin(__builtin_amdgcn_s_wait_asynccnt)
#define WAIT_ASYNC(n) __builtin_amdgcn_s_wait_asynccnt(n)
#else
#define WAIT_ASYNC(n) asm volatile("s_wait_asynccnt %0" ::"i"(n) : "memory")
#endif

// builtin takes (AS1 int4*, AS3 int4*, int imm_offset, int cpol) with
// GCC-style vector_size(16) int pointee (per hipcc diagnostic).
typedef int vsi4 __attribute__((vector_size(16)));
typedef __attribute__((address_space(1))) vsi4* as1_v4i;
typedef __attribute__((address_space(3))) vsi4* as3_v4i;
#define ASYNC_CP16(gsrc, ldst) \
  __builtin_amdgcn_global_load_async_to_lds_b128((as1_v4i)(gsrc), (as3_v4i)(ldst), 0, 0)

union FragBF { v16bf v; uint4 u[2]; };

__device__ __forceinline__ float gelu_exact(float x) {
  return 0.5f * x * (1.0f + erff(x * 0.70710678118654752440f));
}

__device__ __forceinline__ float block_reduce_sum(float v, float* red) {
  int t = threadIdx.x;
  red[t] = v;
  __syncthreads();
  for (int s = 128; s > 0; s >>= 1) {
    if (t < s) red[t] += red[t + s];
    __syncthreads();
  }
  float r = red[0];
  __syncthreads();
  return r;
}

// ---------------------------------------------------------------------------
// f32 -> bf16 conversion (weights)
// ---------------------------------------------------------------------------
__global__ __launch_bounds__(256)
void k_cvt_bf16(const float* __restrict__ src, bf16_t* __restrict__ dst, int n4) {
  int i = blockIdx.x * 256 + threadIdx.x;
  if (i < n4) {
    float4 f = ((const float4*)src)[i];
    bf16_t* d = dst + (size_t)i * 4;
    d[0] = (bf16_t)f.x; d[1] = (bf16_t)f.y; d[2] = (bf16_t)f.z; d[3] = (bf16_t)f.w;
  }
}

// ---------------------------------------------------------------------------
// LayerNorm over E=1024. One 256-thread block per row (4 floats / thread).
// srow = (r / groupRows)*groupStride + r%groupRows + extra  handles the
// inputs[:, :-1] / inputs[:, -1] views without a gather pass.
// ---------------------------------------------------------------------------
__global__ __launch_bounds__(256)
void k_ln(const float* __restrict__ src, const float* __restrict__ g,
          const float* __restrict__ b, bf16_t* __restrict__ obf,
          float* __restrict__ of32, int groupRows, int groupStride, int extra) {
  __shared__ float red[256];
  int r = blockIdx.x;
  size_t srow = (size_t)(r / groupRows) * groupStride + (r % groupRows) + extra;
  const float* xr = src + srow * 1024;
  int t = threadIdx.x;
  float4 x = ((const float4*)xr)[t];
  float mean = block_reduce_sum(x.x + x.y + x.z + x.w, red) * (1.f / 1024.f);
  float d0 = x.x - mean, d1 = x.y - mean, d2 = x.z - mean, d3 = x.w - mean;
  float var = block_reduce_sum(d0*d0 + d1*d1 + d2*d2 + d3*d3, red) * (1.f / 1024.f);
  float rstd = rsqrtf(var + LN_EPS);
  float4 gg = ((const float4*)g)[t];
  float4 bb = ((const float4*)b)[t];
  float y0 = d0 * rstd * gg.x + bb.x;
  float y1 = d1 * rstd * gg.y + bb.y;
  float y2 = d2 * rstd * gg.z + bb.z;
  float y3 = d3 * rstd * gg.w + bb.w;
  size_t o = (size_t)r * 1024 + (size_t)t * 4;
  obf[o + 0] = (bf16_t)y0; obf[o + 1] = (bf16_t)y1;
  obf[o + 2] = (bf16_t)y2; obf[o + 3] = (bf16_t)y3;
  if (of32) { of32[o+0] = y0; of32[o+1] = y1; of32[o+2] = y2; of32[o+3] = y3; }
}

// ---------------------------------------------------------------------------
// Fused res_ln -> ln2 (two chained LayerNorms), bf16 out for next GEMM.
// ---------------------------------------------------------------------------
__global__ __launch_bounds__(256)
void k_ln_double(const float* __restrict__ x, const float* __restrict__ g1,
                 const float* __restrict__ b1, const float* __restrict__ g2,
                 const float* __restrict__ b2, bf16_t* __restrict__ out) {
  __shared__ float red[256];
  size_t row = blockIdx.x;
  const float* xr = x + row * 1024;
  int t = threadIdx.x;
  float4 v = ((const float4*)xr)[t];
  float mean = block_reduce_sum(v.x + v.y + v.z + v.w, red) * (1.f / 1024.f);
  float d0 = v.x - mean, d1 = v.y - mean, d2 = v.z - mean, d3 = v.w - mean;
  float var = block_reduce_sum(d0*d0 + d1*d1 + d2*d2 + d3*d3, red) * (1.f / 1024.f);
  float rstd = rsqrtf(var + LN_EPS);
  float4 gg = ((const float4*)g1)[t];
  float4 bb = ((const float4*)b1)[t];
  float y0 = d0 * rstd * gg.x + bb.x;
  float y1 = d1 * rstd * gg.y + bb.y;
  float y2 = d2 * rstd * gg.z + bb.z;
  float y3 = d3 * rstd * gg.w + bb.w;
  float mean2 = block_reduce_sum(y0 + y1 + y2 + y3, red) * (1.f / 1024.f);
  float e0 = y0 - mean2, e1 = y1 - mean2, e2 = y2 - mean2, e3 = y3 - mean2;
  float var2 = block_reduce_sum(e0*e0 + e1*e1 + e2*e2 + e3*e3, red) * (1.f / 1024.f);
  float rstd2 = rsqrtf(var2 + LN_EPS);
  float4 g2v = ((const float4*)g2)[t];
  float4 b2v = ((const float4*)b2)[t];
  bf16_t* orow = out + row * 1024 + (size_t)t * 4;
  orow[0] = (bf16_t)(e0 * rstd2 * g2v.x + b2v.x);
  orow[1] = (bf16_t)(e1 * rstd2 * g2v.y + b2v.y);
  orow[2] = (bf16_t)(e2 * rstd2 * g2v.z + b2v.z);
  orow[3] = (bf16_t)(e3 * rstd2 * g2v.w + b2v.w);
}

// ---------------------------------------------------------------------------
// Tiled WMMA GEMM:  C[M,N] = A[M,K] @ W[N,K]^T  (+ bias, epilogue variants)
// Block tile 128x128, 8 waves (2x4), wave tile 64x32 -> 4x2 accum fragments.
// LDS rows padded to 48 bf16 (96B, 16B-aligned) so fragment reads are
// ds_load_b128. Fragment element mapping used here (A and B share one k order):
//   A (16x32): lane m = l&15, element j -> K = j + 16*(l>>4)
//   B (32x16): lane n = l&15, element j -> K = j + 16*(l>>4)
//   C (16x16): VGPR r, lane   -> row = r + 8*(l>>4), col = l&15
// Tile staging: async global->LDS copies (ASYNCcnt) with double buffering;
// each thread owns 2 x 16B chunks per matrix per K-step (4 async ops), so
// s_wait_asynccnt<=4 guarantees the current tile is resident while the next
// tile's copies stay in flight.
// ---------------------------------------------------------------------------
enum EpKind { EP_BF16 = 0, EP_GELU_BF16 = 1, EP_F32 = 2, EP_RES_F32 = 3 };

template <int EP>
__global__ __launch_bounds__(256)
void k_gemm(const bf16_t* __restrict__ A, const bf16_t* __restrict__ W,
            const float* __restrict__ bias, const float* __restrict__ res,
            void* __restrict__ Cout, int Mrows, int N, int K) {
#if USE_ASYNC_LDS
  __shared__ bf16_t As[2][128][48];
  __shared__ bf16_t Bs[2][128][48];
#else
  __shared__ bf16_t As[1][128][48];
  __shared__ bf16_t Bs[1][128][48];
#endif
  const int tid = threadIdx.x;
  const int wave = tid >> 5, lane = tid & 31;
  const int half = lane >> 4, l15 = lane & 15;
  const int wm = (wave >> 2) * 64;   // wave row offset in block tile
  const int wn = (wave & 3) * 32;    // wave col offset in block tile
  const int bm = blockIdx.y * 128;
  const int bn = blockIdx.x * 128;

  const v8f vzero = {0.f, 0.f, 0.f, 0.f, 0.f, 0.f, 0.f, 0.f};
  v8f acc[4][2];
#pragma unroll
  for (int mi = 0; mi < 4; ++mi)
#pragma unroll
    for (int nj = 0; nj < 2; ++nj) acc[mi][nj] = vzero;

  // per-thread staging chunk coordinates (2 x 16B chunks per matrix)
  const int r0 = (tid * 2) >> 2, kc0 = ((tid * 2) & 3) * 8;
  const int r1 = (tid * 2 + 1) >> 2, kc1 = ((tid * 2 + 1) & 3) * 8;

#if USE_ASYNC_LDS
  // clamp padded A rows (their C rows are never stored)
  const int ga0 = (bm + r0 < Mrows) ? (bm + r0) : (Mrows - 1);
  const int ga1 = (bm + r1 < Mrows) ? (bm + r1) : (Mrows - 1);
  auto issue = [&](int buf, int kk) {
    ASYNC_CP16(A + (size_t)ga0 * K + kk + kc0, &As[buf][r0][kc0]);
    ASYNC_CP16(A + (size_t)ga1 * K + kk + kc1, &As[buf][r1][kc1]);
    ASYNC_CP16(W + (size_t)(bn + r0) * K + kk + kc0, &Bs[buf][r0][kc0]);
    ASYNC_CP16(W + (size_t)(bn + r1) * K + kk + kc1, &Bs[buf][r1][kc1]);
  };
  issue(0, 0);
#endif

  for (int kk = 0; kk < K; kk += 32) {
#if USE_ASYNC_LDS
    const int buf = (kk >> 5) & 1;
    if (kk + 32 < K) {
      issue(buf ^ 1, kk + 32);   // prefetch next tile into the other buffer
      WAIT_ASYNC(4);             // current tile resident; next stays in flight
    } else {
      WAIT_ASYNC(0);
    }
#else
    const int buf = 0;
    {
      uint4 aval = make_uint4(0u, 0u, 0u, 0u);
      if (bm + r0 < Mrows) aval = *(const uint4*)(A + (size_t)(bm + r0) * K + kk + kc0);
      *(uint4*)(&As[0][r0][kc0]) = aval;
      uint4 aval1 = make_uint4(0u, 0u, 0u, 0u);
      if (bm + r1 < Mrows) aval1 = *(const uint4*)(A + (size_t)(bm + r1) * K + kk + kc1);
      *(uint4*)(&As[0][r1][kc1]) = aval1;
      *(uint4*)(&Bs[0][r0][kc0]) = *(const uint4*)(W + (size_t)(bn + r0) * K + kk + kc0);
      *(uint4*)(&Bs[0][r1][kc1]) = *(const uint4*)(W + (size_t)(bn + r1) * K + kk + kc1);
      if (kk + 32 < K) {
        if (bm + r0 < Mrows) __builtin_prefetch(A + (size_t)(bm + r0) * K + kk + 32 + kc0, 0, 1);
        __builtin_prefetch(W + (size_t)(bn + r0) * K + kk + 32 + kc0, 0, 1);
      }
    }
#endif
    __syncthreads();
#pragma unroll
    for (int mi = 0; mi < 4; ++mi) {
      FragBF a;
      const bf16_t* ap = &As[buf][wm + mi * 16 + l15][0];
      a.u[0] = *(const uint4*)(ap + 16 * half);
      a.u[1] = *(const uint4*)(ap + 16 * half + 8);
#pragma unroll
      for (int nj = 0; nj < 2; ++nj) {
        FragBF bfr;
        const bf16_t* bp = &Bs[buf][wn + nj * 16 + l15][16 * half];
        bfr.u[0] = *(const uint4*)(bp);
        bfr.u[1] = *(const uint4*)(bp + 8);
        acc[mi][nj] = __builtin_amdgcn_wmma_f32_16x16x32_bf16(
            false, a.v, false, bfr.v, (short)0, acc[mi][nj], false, false);
      }
    }
    __syncthreads();
  }

#pragma unroll
  for (int mi = 0; mi < 4; ++mi)
#pragma unroll
    for (int nj = 0; nj < 2; ++nj)
#pragma unroll
      for (int r = 0; r < 8; ++r) {
        int grow = bm + wm + mi * 16 + r + 8 * half;
        int gcol = bn + wn + nj * 16 + l15;
        if (grow < Mrows) {
          float x = acc[mi][nj][r] + bias[gcol];
          if (EP == EP_GELU_BF16) x = gelu_exact(x);
          if (EP == EP_RES_F32) x += res[(size_t)grow * N + gcol];
          size_t o = (size_t)grow * N + gcol;
          if (EP == EP_F32 || EP == EP_RES_F32) ((float*)Cout)[o] = x;
          else ((bf16_t*)Cout)[o] = (bf16_t)x;
        }
      }
}

// ---------------------------------------------------------------------------
// Fused attention: one block per (query row-block of 32, head, b*T+t).
// NF=196 padded to 224 (7 K-steps of 32). S kept in f32 LDS; softmax uses
// __shfl_xor wave reductions; P converted bf16 on the fly for the PV WMMA.
// ---------------------------------------------------------------------------
__global__ __launch_bounds__(256)
void k_attn(const bf16_t* __restrict__ q, const bf16_t* __restrict__ k,
            const bf16_t* __restrict__ v, bf16_t* __restrict__ o) {
  constexpr int NF = 196, NFP = 224, E = 1024, D = 64, T = 15;
  __shared__ bf16_t Qs[32][64];       // 32 query rows x 64 dims
  __shared__ bf16_t KVs[NFP][64];     // K, later reused for V
  __shared__ float Ss[32][232];       // scores / probabilities

  const int tid = threadIdx.x;
  const int wave = tid >> 5, lane = tid & 31;
  const int half = lane >> 4, l15 = lane & 15;
  const int rb = blockIdx.x;          // 0..6
  const int h = blockIdx.y;           // 0..15
  const int bt = blockIdx.z;          // b*T + t
  const int b = bt / T;
  const v8f vzero = {0.f, 0.f, 0.f, 0.f, 0.f, 0.f, 0.f, 0.f};

  { // load Q tile (zero-pad rows >= NF)
    int row = tid >> 3, c8 = (tid & 7) * 8;
    int qrow = rb * 32 + row;
    uint4 val = make_uint4(0u, 0u, 0u, 0u);
    if (qrow < NF)
      val = *(const uint4*)(q + (size_t)(b * NF + qrow) * E + h * D + c8);
    *(uint4*)(&Qs[row][c8]) = val;
  }
  for (int id = tid; id < NFP * 8; id += 256) { // load K tile (zero-padded)
    int row = id >> 3, c8 = (id & 7) * 8;
    uint4 val = make_uint4(0u, 0u, 0u, 0u);
    if (row < NF)
      val = *(const uint4*)(k + (size_t)(bt * NF + row) * E + h * D + c8);
    *(uint4*)(&KVs[row][c8]) = val;
  }
  __syncthreads();

  // S = (Q K^T) / sqrt(D) : 2 x 14 tiles of 16x16, K-dim = 64
  for (int id = wave; id < 28; id += 8) {
    int mi = id / 14, nj = id % 14;
    v8f acc = vzero;
#pragma unroll
    for (int kk = 0; kk < 64; kk += 32) {
      FragBF a;
      const bf16_t* ap = &Qs[mi * 16 + l15][kk];
      a.u[0] = *(const uint4*)(ap + 16 * half);
      a.u[1] = *(const uint4*)(ap + 16 * half + 8);
      FragBF bfr;
      const bf16_t* bp = &KVs[nj * 16 + l15][kk + 16 * half];
      bfr.u[0] = *(const uint4*)(bp);
      bfr.u[1] = *(const uint4*)(bp + 8);
      acc = __builtin_amdgcn_wmma_f32_16x16x32_bf16(
          false, a.v, false, bfr.v, (short)0, acc, false, false);
    }
#pragma unroll
    for (int r = 0; r < 8; ++r)
      Ss[mi * 16 + r + 8 * half][nj * 16 + l15] = acc[r] * 0.125f; // 1/sqrt(64)
  }
  __syncthreads();

  // overlap: load V tile into KVs (K no longer needed) ...
  for (int id = tid; id < NFP * 8; id += 256) {
    int row = id >> 3, c8 = (id & 7) * 8;
    uint4 val = make_uint4(0u, 0u, 0u, 0u);
    if (row < NF)
      val = *(const uint4*)(v + (size_t)(bt * NF + row) * E + h * D + c8);
    *(uint4*)(&KVs[row][c8]) = val;
  }
  // ... while doing the softmax (4 rows per wave, lane-parallel columns)
  for (int i = 0; i < 4; ++i) {
    int r = wave * 4 + i;
    float mx = -1e30f;
    for (int c = lane; c < NF; c += 32) mx = fmaxf(mx, Ss[r][c]);
    for (int off = 16; off > 0; off >>= 1) mx = fmaxf(mx, __shfl_xor(mx, off, 32));
    float sum = 0.f;
    for (int c = lane; c < NF; c += 32) {
      float e = __expf(Ss[r][c] - mx);
      Ss[r][c] = e;
      sum += e;
    }
    for (int off = 16; off > 0; off >>= 1) sum += __shfl_xor(sum, off, 32);
    float inv = 1.f / sum;
    for (int c = lane; c < NF; c += 32) Ss[r][c] *= inv;
    for (int c = NF + lane; c < NFP; c += 32) Ss[r][c] = 0.f; // padded keys
  }
  __syncthreads();

  // O = P V : 2 x 4 tiles, one per wave, K-dim = 224
  {
    int mi = wave >> 2, nj = wave & 3;
    v8f acc = vzero;
    for (int kk = 0; kk < NFP; kk += 32) {
      v16bf a;
#pragma unroll
      for (int j = 0; j < 16; ++j) {
        int kx = kk + j + 16 * half;
        a[j] = (bf16_t)Ss[mi * 16 + l15][kx];   // on-the-fly f32 -> bf16
      }
      v16bf bv;
#pragma unroll
      for (int j = 0; j < 16; ++j)
        bv[j] = KVs[kk + j + 16 * half][nj * 16 + l15];
      acc = __builtin_amdgcn_wmma_f32_16x16x32_bf16(
          false, a, false, bv, (short)0, acc, false, false);
    }
#pragma unroll
    for (int r = 0; r < 8; ++r) {
      int qrow = rb * 32 + mi * 16 + r + 8 * half;
      if (qrow < NF)
        o[(size_t)(bt * NF + qrow) * E + h * D + nj * 16 + l15] = (bf16_t)acc[r];
    }
  }
}

// ---------------------------------------------------------------------------
// Orchestration
// ---------------------------------------------------------------------------
extern "C" void kernel_launch(void* const* d_in, const int* in_sizes, int n_in,
                              void* d_out, int out_size, void* d_ws, size_t ws_size,
                              hipStream_t stream) {
  (void)in_sizes; (void)n_in; (void)out_size; (void)ws_size;
  const float* inputs     = (const float*)d_in[0];
  const float* in_proj_w  = (const float*)d_in[1];
  const float* in_proj_b  = (const float*)d_in[2];
  const float* out_proj_w = (const float*)d_in[3];
  const float* out_proj_b = (const float*)d_in[4];
  const float* ln_q_g  = (const float*)d_in[5];
  const float* ln_q_b  = (const float*)d_in[6];
  const float* ln_kv_g = (const float*)d_in[7];
  const float* ln_kv_b = (const float*)d_in[8];
  const float* mlpq_w1 = (const float*)d_in[9];
  const float* mlpq_b1 = (const float*)d_in[10];
  const float* mlpq_w2 = (const float*)d_in[11];
  const float* mlpq_b2 = (const float*)d_in[12];
  const float* res_ln_g = (const float*)d_in[13];
  const float* res_ln_b = (const float*)d_in[14];
  const float* ln2_g = (const float*)d_in[15];
  const float* ln2_b = (const float*)d_in[16];
  const float* mlp_w1 = (const float*)d_in[17];
  const float* mlp_b1 = (const float*)d_in[18];
  const float* mlp_w2 = (const float*)d_in[19];
  const float* mlp_b2 = (const float*)d_in[20];

  const int Bv = 8, T = 15, NF = 196, E = 1024, M = 4096;
  const int RQ = Bv * NF;        // 1568 query rows
  const int RKV = Bv * T * NF;   // 23520 kv rows

  char* base = (char*)d_ws;
  size_t off = 0;
  auto alloc = [&](size_t bytes) -> void* {
    void* p = base + off;
    off += (bytes + 255) & ~(size_t)255;
    return p;
  };
  bf16_t* w_inproj = (bf16_t*)alloc((size_t)3 * E * E * 2);
  bf16_t* w_out    = (bf16_t*)alloc((size_t)E * E * 2);
  bf16_t* w_q1     = (bf16_t*)alloc((size_t)M * E * 2);
  bf16_t* w_q2     = (bf16_t*)alloc((size_t)E * M * 2);
  bf16_t* w_m1     = (bf16_t*)alloc((size_t)M * E * 2);
  bf16_t* w_m2     = (bf16_t*)alloc((size_t)E * M * 2);
  bf16_t* qln_bf   = (bf16_t*)alloc((size_t)RQ * E * 2);
  float*  kvln_f   = (float*) alloc((size_t)RKV * E * 4);
  bf16_t* kvln_bf  = (bf16_t*)alloc((size_t)RKV * E * 2);
  bf16_t* qbuf     = (bf16_t*)alloc((size_t)RQ * E * 2);
  bf16_t* kbuf     = (bf16_t*)alloc((size_t)RKV * E * 2);
  bf16_t* vbuf     = (bf16_t*)alloc((size_t)RKV * E * 2);
  bf16_t* attn_bf  = (bf16_t*)alloc((size_t)RKV * E * 2);
  bf16_t* attended = (bf16_t*)alloc((size_t)RKV * E * 2);
  bf16_t* hidden   = (bf16_t*)alloc((size_t)RKV * M * 2);
  float*  mlpqout  = (float*) alloc((size_t)RKV * E * 4);
  bf16_t* x2_bf    = (bf16_t*)alloc((size_t)RKV * E * 2);

  auto cvt = [&](const float* s, bf16_t* d, size_t n) {
    k_cvt_bf16<<<dim3((unsigned)((n / 4 + 255) / 256)), dim3(256), 0, stream>>>(
        s, d, (int)(n / 4));
  };
  cvt(in_proj_w, w_inproj, (size_t)3 * E * E);
  cvt(out_proj_w, w_out, (size_t)E * E);
  cvt(mlpq_w1, w_q1, (size_t)M * E);
  cvt(mlpq_w2, w_q2, (size_t)E * M);
  cvt(mlp_w1, w_m1, (size_t)M * E);
  cvt(mlp_w2, w_m2, (size_t)E * M);

  // LayerNorms over the input views (kv: rows (b,t,n) of inputs[:, :-1];
  // q: rows (b,n) of inputs[:, -1]).
  k_ln<<<dim3(RKV), 256, 0, stream>>>(inputs, ln_kv_g, ln_kv_b, kvln_bf, kvln_f,
                                      T * NF, 16 * NF, 0);
  k_ln<<<dim3(RQ), 256, 0, stream>>>(inputs, ln_q_g, ln_q_b, qln_bf,
                                     (float*)nullptr, NF, 16 * NF, T * NF);

  auto grid = [&](int n, int m) {
    return dim3((unsigned)(n / 128), (unsigned)((m + 127) / 128));
  };

  // QKV projections
  k_gemm<EP_BF16><<<grid(E, RQ), 256, 0, stream>>>(
      qln_bf, w_inproj, in_proj_b, nullptr, qbuf, RQ, E, E);
  k_gemm<EP_BF16><<<grid(E, RKV), 256, 0, stream>>>(
      kvln_bf, w_inproj + (size_t)E * E, in_proj_b + E, nullptr, kbuf, RKV, E, E);
  k_gemm<EP_BF16><<<grid(E, RKV), 256, 0, stream>>>(
      kvln_bf, w_inproj + (size_t)2 * E * E, in_proj_b + 2 * E, nullptr, vbuf,
      RKV, E, E);

  // fused attention per (row-block, head, b*t)
  k_attn<<<dim3(7, 16, Bv * T), 256, 0, stream>>>(qbuf, kbuf, vbuf, attn_bf);

  // out projection
  k_gemm<EP_BF16><<<grid(E, RKV), 256, 0, stream>>>(
      attn_bf, w_out, out_proj_b, nullptr, attended, RKV, E, E);

  // mlpq: fc1 (+GELU) then fc2 (+bias +kv_ln residual, f32 out)
  k_gemm<EP_GELU_BF16><<<grid(M, RKV), 256, 0, stream>>>(
      attended, w_q1, mlpq_b1, nullptr, hidden, RKV, M, E);
  k_gemm<EP_RES_F32><<<grid(E, RKV), 256, 0, stream>>>(
      hidden, w_q2, mlpq_b2, kvln_f, mlpqout, RKV, E, M);

  // res_ln then ln2, fused
  k_ln_double<<<dim3(RKV), 256, 0, stream>>>(mlpqout, res_ln_g, res_ln_b, ln2_g,
                                             ln2_b, x2_bf);

  // final MLP: fc1 (+GELU) then fc2 -> f32 output
  k_gemm<EP_GELU_BF16><<<grid(M, RKV), 256, 0, stream>>>(
      x2_bf, w_m1, mlp_b1, nullptr, hidden, RKV, M, E);
  k_gemm<EP_F32><<<grid(E, RKV), 256, 0, stream>>>(
      hidden, w_m2, mlp_b2, nullptr, (float*)d_out, RKV, E, M);
}
